// FootballGCN_11072425689238
// MI455X (gfx1250) — matrix operand order repaired and checked by reading, not compiled, not run on hardware
//
#include <hip/hip_runtime.h>
#include <hip/hip_bf16.h>

typedef __attribute__((ext_vector_type(16))) _Float16 v16h;
typedef __attribute__((ext_vector_type(8)))  _Float16 v8h;
typedef __attribute__((ext_vector_type(8)))  float    v8f;

#define GHID 64

// ---------------- elementwise helpers ----------------
__global__ void k_fill(float* __restrict__ p, float v, long n) {
    long i = (long)blockIdx.x * blockDim.x + threadIdx.x;
    if (i < n) p[i] = v;
}

__global__ void k_count_deg(const int* __restrict__ dst, float* __restrict__ deg, int E) {
    int e = blockIdx.x * blockDim.x + threadIdx.x;
    if (e < E) atomicAdd(&deg[dst[e]], 1.0f);
}

__global__ void k_rsqrt_inplace(float* __restrict__ d, int n) {
    int i = blockIdx.x * blockDim.x + threadIdx.x;
    if (i < n) d[i] = rsqrtf(d[i]);   // deg >= 1 due to self-loop
}

// agg = gemm_out * dis^2   (self-loop contribution seeds the scatter buffer)
__global__ void k_self_init(const float* __restrict__ g, const float* __restrict__ dis,
                            float* __restrict__ agg, long n64) {
    long i = (long)blockIdx.x * blockDim.x + threadIdx.x;
    if (i >= n64) return;
    float di = dis[i >> 6];
    agg[i] = g[i] * di * di;
}

// 32 lanes per edge, 2 channels per lane -> 64 channels
__global__ void k_edge_scatter(const int* __restrict__ src, const int* __restrict__ dst,
                               const float* __restrict__ dis,
                               const float* __restrict__ g, float* __restrict__ agg, int E) {
    long tid = (long)blockIdx.x * blockDim.x + threadIdx.x;
    long e = tid >> 5;
    if (e >= E) return;
    int c0 = ((int)tid & 31) * 2;
    int s = src[e], d = dst[e];
    float w = dis[s] * dis[d];
    float2 v = *(const float2*)(g + (long)s * GHID + c0);
    atomicAdd(&agg[(long)d * GHID + c0],     w * v.x);
    atomicAdd(&agg[(long)d * GHID + c0 + 1], w * v.y);
}

__global__ void k_bias_relu(const float* __restrict__ agg, const float* __restrict__ bias,
                            float* __restrict__ out, long n64) {
    long i = (long)blockIdx.x * blockDim.x + threadIdx.x;
    if (i >= n64) return;
    out[i] = fmaxf(agg[i] + bias[i & 63], 0.0f);
}

__global__ void k_pool_scatter(const float* __restrict__ h, const int* __restrict__ batch,
                               float* __restrict__ pool, float* __restrict__ cnt, long n64) {
    long idx = (long)blockIdx.x * blockDim.x + threadIdx.x;
    if (idx >= n64) return;
    int i = (int)(idx >> 6), c = (int)(idx & 63);
    int b = batch[i];
    atomicAdd(&pool[(long)b * GHID + c], h[idx]);
    if (c == 0) atomicAdd(&cnt[b], 1.0f);
}

// ---------------- WMMA GEMM: C(Mx64) = A(MxK) @ W(Kx64), f16 in / f32 acc ----------------
// One wave per 16x64 output stripe: A fragment loaded once per K-chunk, reused by
// 4 independent WMMAs (independent accumulators -> back-to-back XDL issue, no hazards).
// W is staged in LDS as f16, transposed (column-major, stride K+8 halves => lanes 0..15
// hit banks 0,4,8,...,60 with 4 consecutive words each: conflict-free ds_load_b128).
template<int K>
__global__ void k_gcn_gemm(const float* __restrict__ A, const float* __restrict__ W,
                           float* __restrict__ C, int M) {
    constexpr int LDW = K + 8;                    // padded LDS stride in halves
    __shared__ _Float16 WT[GHID * LDW];

    // cooperative transpose-convert of W (K x 64, f32) into WT (64 x K, f16)
    for (int idx = threadIdx.x; idx < GHID * K; idx += blockDim.x) {
        int c = idx & (GHID - 1);
        int k = idx >> 6;
        WT[c * LDW + k] = (_Float16)W[k * GHID + c];
    }
    __syncthreads();                              // every thread reaches the barrier

    const int wave = blockIdx.x * (blockDim.x >> 5) + (threadIdx.x >> 5);
    if (wave * 16 >= M) return;                   // wave-uniform: EXEC all-1s for WMMA

    const int lane  = threadIdx.x & 31;
    const int laneM = lane & 15;
    const int hi    = lane >> 4;                  // 0: lanes 0-15, 1: lanes 16-31
    const float* __restrict__ arow = A + (long)(wave * 16 + laneM) * K;

    v8f acc[4] = {};
#pragma unroll
    for (int kb = 0; kb < K; kb += 32) {
        // A 16x32 f16 fragment: half h -> K = kb + 16*hi + h
        v16h a;
#pragma unroll
        for (int i = 0; i < 8; ++i) {
            a[i]     = (_Float16)arow[kb + hi * 16 + i];
            a[i + 8] = (_Float16)arow[kb + hi * 16 + 8 + i];
        }
#pragma unroll
        for (int t = 0; t < 4; ++t) {
            // B 32x16 fragment: half h -> K = kb + 16*hi + h, col = t*16 + (lane&15)
            const int col = t * 16 + laneM;
            const _Float16* bp = &WT[col * LDW + kb + hi * 16];  // 16B-aligned, contiguous
            v8h blo = *(const v8h*)(bp);
            v8h bhi = *(const v8h*)(bp + 8);
            v16h b;
#pragma unroll
            for (int i = 0; i < 8; ++i) { b[i] = blo[i]; b[i + 8] = bhi[i]; }
            acc[t] = __builtin_amdgcn_wmma_f32_16x16x32_f16(
                false, a, false, b, (short)0, acc[t], false, false);
        }
    }
    // C f32 layout: VGPR r -> row = r + 8*hi, col = lane&15 (per tile t)
#pragma unroll
    for (int t = 0; t < 4; ++t) {
        float* cp = C + (long)(wave * 16 + hi * 8) * GHID + t * 16 + laneM;
#pragma unroll
        for (int r = 0; r < 8; ++r)
            cp[(long)r * GHID] = acc[t][r];
    }
}

// ---------------- tiny MLP head: (B x 64) -> relu(64->32) -> 32->1 ----------------
__global__ void k_mlp(const float* __restrict__ pool, const float* __restrict__ cnt,
                      const float* __restrict__ Wm1, const float* __restrict__ bm1,
                      const float* __restrict__ Wm2, const float* __restrict__ bm2,
                      float* __restrict__ out, int B) {
    int b = blockIdx.x * blockDim.x + threadIdx.x;
    if (b >= B) return;
    float inv = 1.0f / fmaxf(cnt[b], 1.0f);
    float p[GHID];
#pragma unroll
    for (int i = 0; i < GHID; ++i) p[i] = pool[(long)b * GHID + i] * inv;
    float acc = bm2[0];
    for (int j = 0; j < 32; ++j) {
        float h = bm1[j];
#pragma unroll
        for (int i = 0; i < GHID; ++i) h += p[i] * Wm1[i * 32 + j];
        acc += fmaxf(h, 0.0f) * Wm2[j];
    }
    out[b] = acc;
}

// ---------------- driver ----------------
extern "C" void kernel_launch(void* const* d_in, const int* in_sizes, int n_in,
                              void* d_out, int out_size, void* d_ws, size_t ws_size,
                              hipStream_t stream) {
    const float* x    = (const float*)d_in[0];
    const int*   ei   = (const int*)d_in[1];     // (2, E): row0 = src, row1 = dst
    const int*   batch= (const int*)d_in[2];
    const float* W0   = (const float*)d_in[3];  const float* b0  = (const float*)d_in[4];
    const float* W1   = (const float*)d_in[5];  const float* b1  = (const float*)d_in[6];
    const float* W2   = (const float*)d_in[7];  const float* b2  = (const float*)d_in[8];
    const float* Wm1  = (const float*)d_in[9];  const float* bm1 = (const float*)d_in[10];
    const float* Wm2  = (const float*)d_in[11]; const float* bm2 = (const float*)d_in[12];
    float* out = (float*)d_out;

    const int N = in_sizes[0] / 128;   // IN_C = 128
    const int E = in_sizes[1] / 2;
    const int B = out_size;            // OUT_C = 1
    const long n64 = (long)N * GHID;

    float* buf0 = (float*)d_ws;            // GEMM output
    float* buf1 = buf0 + n64;              // aggregation buffer
    float* buf2 = buf1 + n64;              // activated hidden state
    float* dis  = buf2 + n64;              // deg -> deg^-1/2 (N)
    float* pool = dis + N;                 // (B x 64), cnt appended right after
    float* cnt  = pool + (long)B * GHID;   // (B)

    const int* src = ei;
    const int* dst = ei + E;
    const int T = 256;

    // degrees (self-loop seeds deg=1), then deg^-1/2 in place
    k_fill<<<(N + T - 1) / T, T, 0, stream>>>(dis, 1.0f, (long)N);
    k_count_deg<<<(E + T - 1) / T, T, 0, stream>>>(dst, dis, E);
    k_rsqrt_inplace<<<(N + T - 1) / T, T, 0, stream>>>(dis, N);

    const int stripes = N / 16;                   // one wave per 16x64 stripe
    const int gemmBlocks = (stripes + 7) / 8;     // 8 waves per 256-thread block
    const long eThreads = (long)E * 32;
    const int eBlocks = (int)((eThreads + T - 1) / T);
    const int nBlocks = (int)((n64 + T - 1) / T);

    // layer 0: K = 128
    k_gcn_gemm<128><<<gemmBlocks, T, 0, stream>>>(x, W0, buf0, N);
    k_self_init<<<nBlocks, T, 0, stream>>>(buf0, dis, buf1, n64);
    k_edge_scatter<<<eBlocks, T, 0, stream>>>(src, dst, dis, buf0, buf1, E);
    k_bias_relu<<<nBlocks, T, 0, stream>>>(buf1, b0, buf2, n64);

    // layers 1,2: K = 64
    const float* Ws[2]     = {W1, W2};
    const float* biases[2] = {b1, b2};
    for (int l = 0; l < 2; ++l) {
        k_gcn_gemm<64><<<gemmBlocks, T, 0, stream>>>(buf2, Ws[l], buf0, N);
        k_self_init<<<nBlocks, T, 0, stream>>>(buf0, dis, buf1, n64);
        k_edge_scatter<<<eBlocks, T, 0, stream>>>(src, dst, dis, buf0, buf1, E);
        k_bias_relu<<<nBlocks, T, 0, stream>>>(buf1, biases[l], buf2, n64);
    }

    // global mean pool + MLP head
    long poolN = (long)B * GHID + B;              // pool and cnt are contiguous
    k_fill<<<(int)((poolN + T - 1) / T), T, 0, stream>>>(pool, 0.0f, poolN);
    k_pool_scatter<<<nBlocks, T, 0, stream>>>(buf2, batch, pool, cnt, n64);
    k_mlp<<<(B + 127) / 128, 128, 0, stream>>>(pool, cnt, Wm1, bm1, Wm2, bm2, out, B);
}
